// SAPP_17179869888
// MI455X (gfx1250) — matrix operand order repaired and checked by reading, not compiled, not run on hardware
//
#include <hip/hip_runtime.h>

// ---------------------------------------------------------------------------
// MI455X (gfx1250) kNN + edge-MLP + max-pool.
//  * whole problem is L2-resident (xyz 0.4MB, out 16.8MB vs 192MB L2)
//  * kNN: batch points (48KB) staged to LDS via global_load_async_to_lds_b128
//    (ASYNCcnt), register-resident branchless top-16, b128 broadcast scan
//  * MLP: v_wmma_f32_16x16x32_f16; weights pre-swizzled to B-frag order in LDS;
//    wave-private LDS staging ordered with s_wait_dscnt (LDS in-order per wave)
//  * output staged in LDS tile and written back fully coalesced
// ---------------------------------------------------------------------------

typedef __attribute__((ext_vector_type(16))) _Float16 v16h;
typedef __attribute__((ext_vector_type(8)))  _Float16 v8h;
typedef __attribute__((ext_vector_type(8)))  float    v8f;

#define NPTS    4096
#define KNN     16
#define HSTRIDE 72          // halves per LDS activation row (64 + 8 pad)
#define MLPB    256
#define WAVES   8
#define PPW     8

#if defined(__AMDGCN__) && __has_builtin(__builtin_amdgcn_global_load_async_to_lds_b128)
#define HAVE_ASYNC_LDS 1
// signature (from clang diagnostic): (v4i AS1*, v4i AS3*, imm int, imm int)
typedef int v4i_t __attribute__((vector_size(16)));
typedef __attribute__((address_space(1))) v4i_t* gas_p;
typedef __attribute__((address_space(3))) v4i_t* las_p;
#else
#define HAVE_ASYNC_LDS 0
#endif

// Wave-local LDS ordering: LDS ops of one wave complete in order (CDNA5 ISA),
// so waiting DScnt==0 + a compiler memory fence makes cross-lane LDS RAW safe
// without a (block-wide) barrier.
__device__ __forceinline__ void wave_lds_sync()
{
#if defined(__AMDGCN__)
    asm volatile("s_wait_dscnt 0" ::: "memory");
#endif
}

// ------------------------------- kNN kernel --------------------------------
__device__ __forceinline__ void topk_insert(float (&bd)[KNN], int (&bi)[KNN],
                                            float& worst, float d, int m)
{
    // branchless replace-worst; everything stays in VGPRs
    float wv = bd[0];
#pragma unroll
    for (int j = 1; j < KNN; ++j) wv = fmaxf(wv, bd[j]);
    bool done = false;
#pragma unroll
    for (int j = 0; j < KNN; ++j) {
        const bool repl = (!done) && (bd[j] == wv);
        bd[j] = repl ? d : bd[j];
        bi[j] = repl ? m : bi[j];
        done  = done || repl;
    }
    float nw = bd[0];
#pragma unroll
    for (int j = 1; j < KNN; ++j) nw = fmaxf(nw, bd[j]);
    worst = nw;
}

__global__ __launch_bounds__(128) void knn_kernel(const float* __restrict__ xyz,
                                                  int* __restrict__ idx_out)
{
    __shared__ __align__(16) float lx[NPTS];
    __shared__ __align__(16) float ly[NPTS];
    __shared__ __align__(16) float lz[NPTS];

    const int b = blockIdx.x >> 5;                       // 32 blocks per batch
    const int n = ((blockIdx.x & 31) << 7) + threadIdx.x;
    const float* px = xyz + (size_t)b * 3 * NPTS;
    const float* py = px + NPTS;
    const float* pz = py + NPTS;

    // stage whole batch (48KB of the 320KB WGP LDS) — async DMA, no VGPR hop
    for (int i = (threadIdx.x << 2); i < NPTS; i += (128 << 2)) {
#if HAVE_ASYNC_LDS
        __builtin_amdgcn_global_load_async_to_lds_b128((gas_p)(px + i), (las_p)(lx + i), 0, 0);
        __builtin_amdgcn_global_load_async_to_lds_b128((gas_p)(py + i), (las_p)(ly + i), 0, 0);
        __builtin_amdgcn_global_load_async_to_lds_b128((gas_p)(pz + i), (las_p)(lz + i), 0, 0);
#else
        *(float4*)(lx + i) = *(const float4*)(px + i);
        *(float4*)(ly + i) = *(const float4*)(py + i);
        *(float4*)(lz + i) = *(const float4*)(pz + i);
#endif
    }
#if HAVE_ASYNC_LDS
#if __has_builtin(__builtin_amdgcn_s_wait_asynccnt)
    __builtin_amdgcn_s_wait_asynccnt(0);
#else
    asm volatile("s_wait_asynccnt 0" ::: "memory");
#endif
#endif
    __syncthreads();

    const float qx = lx[n], qy = ly[n], qz = lz[n];

    float bd[KNN];
    int   bi[KNN];
#pragma unroll
    for (int j = 0; j < KNN; ++j) { bd[j] = __FLT_MAX__; bi[j] = 0; }
    float worst = __FLT_MAX__;

    // scan 4 candidates per iteration: 3x ds_load_b128 broadcast reads
    for (int m = 0; m < NPTS; m += 4) {
        const float4 vx = *(const float4*)(lx + m);
        const float4 vy = *(const float4*)(ly + m);
        const float4 vz = *(const float4*)(lz + m);
        const float dx0 = vx.x - qx, dy0 = vy.x - qy, dz0 = vz.x - qz;
        const float dx1 = vx.y - qx, dy1 = vy.y - qy, dz1 = vz.y - qz;
        const float dx2 = vx.z - qx, dy2 = vy.z - qy, dz2 = vz.z - qz;
        const float dx3 = vx.w - qx, dy3 = vy.w - qy, dz3 = vz.w - qz;
        const float d0 = fmaf(dx0, dx0, fmaf(dy0, dy0, dz0 * dz0));
        const float d1 = fmaf(dx1, dx1, fmaf(dy1, dy1, dz1 * dz1));
        const float d2 = fmaf(dx2, dx2, fmaf(dy2, dy2, dz2 * dz2));
        const float d3 = fmaf(dx3, dx3, fmaf(dy3, dy3, dz3 * dz3));
        const float dmin = fminf(fminf(d0, d1), fminf(d2, d3));
        if (dmin < worst) {                              // rare path (~K ln N)
            if (d0 < worst) topk_insert(bd, bi, worst, d0, m + 0);
            if (d1 < worst) topk_insert(bd, bi, worst, d1, m + 1);
            if (d2 < worst) topk_insert(bd, bi, worst, d2, m + 2);
            if (d3 < worst) topk_insert(bd, bi, worst, d3, m + 3);
        }
    }

    int* op = idx_out + ((size_t)b * NPTS + n) * KNN;
#pragma unroll
    for (int j = 0; j < KNN; ++j) op[j] = bi[j];
}

// -------------------------- WMMA fragment loaders ---------------------------
// A-fragment (16x32 f16): lanes 0-15 row=lane, K = k0+{0..7, 16..23};
//                         lanes 16-31 same rows, K = k0+{8..15, 24..31}.
__device__ __forceinline__ v16h load_afrag(const _Float16* hrow, int row, int half, int k0)
{
    const _Float16* p = hrow + row * HSTRIDE + k0 + (half ? 8 : 0);
    const v8h lo = *(const v8h*)(p);          // 16B aligned: HSTRIDE*2 % 16 == 0
    const v8h hi = *(const v8h*)(p + 16);
    return __builtin_shufflevector(lo, hi, 0,1,2,3,4,5,6,7,8,9,10,11,12,13,14,15);
}

// B-fragment: staged in LDS in fragment order -> 2 contiguous 16B loads/lane.
__device__ __forceinline__ v16h load_bfrag(const _Float16* sw, int frag, int lane)
{
    const _Float16* p = sw + (((frag << 5) + lane) << 4);
    const v8h lo = *(const v8h*)(p);
    const v8h hi = *(const v8h*)(p + 8);
    return __builtin_shufflevector(lo, hi, 0,1,2,3,4,5,6,7,8,9,10,11,12,13,14,15);
}

// ------------------------------ edge-MLP kernel -----------------------------
__global__ __launch_bounds__(MLPB) void edge_mlp_kernel(
    const float* __restrict__ xyz,
    const float* __restrict__ W0, const float* __restrict__ b0,
    const float* __restrict__ W1, const float* __restrict__ b1,
    const float* __restrict__ W2, const float* __restrict__ b2,
    const int* __restrict__ nbr, float* __restrict__ out)
{
    __shared__ __align__(16) _Float16 sW1[4096];              // 64x64  B-frags  (8KB)
    __shared__ __align__(16) _Float16 sW2[8192];              // 64x128 B-frags (16KB)
    __shared__ __align__(16) _Float16 hAct[WAVES][16 * HSTRIDE]; //             (18KB)
    __shared__ __align__(16) _Float16 oT[128 * 64];           // out tile       (16KB)

    const int tid = threadIdx.x;

    // Stage W1/W2 (f32 -> f16) into B-fragment order:
    // frag id = j*2+k  (j = 16-col tile, k = 32-K step);
// lane l holds col = j*16 + (l&15), K = k*32 + (l&16)/2 + {0..7, 16..23}[elem].
    for (int e = tid; e < 4096; e += MLPB) {
        const int elem = e & 15;
        const int ln   = (e >> 4) & 31;
        const int kj   = e >> 9;                 // 0..7
        const int col  = ((kj >> 1) << 4) + (ln & 15);
        const int kidx = ((kj & 1) << 5) + ((ln & 16) >> 1) + (elem & 7) + ((elem & 8) << 1);
        sW1[e] = (_Float16)W1[col * 64 + kidx];  // B[k][n] = W1[n][k]
    }
    for (int e = tid; e < 8192; e += MLPB) {
        const int elem = e & 15;
        const int ln   = (e >> 4) & 31;
        const int kj   = e >> 9;                 // 0..15
        const int col  = ((kj >> 1) << 4) + (ln & 15);
        const int kidx = ((kj & 1) << 5) + ((ln & 16) >> 1) + (elem & 7) + ((elem & 8) << 1);
        sW2[e] = (_Float16)W2[col * 64 + kidx];
    }
    __syncthreads();

    const int wave = tid >> 5;
    const int lane = tid & 31;
    const int row  = lane & 15;
    const int half = lane >> 4;
    _Float16* hrow = &hAct[wave][0];

    for (int p = 0; p < PPW; ++p) {
        const int pt = (blockIdx.x * WAVES + wave) * PPW + p;  // 0..32767
        const int b  = pt >> 12;
        const int n  = pt & (NPTS - 1);
        const float* px = xyz + (size_t)b * 3 * NPTS;
        __builtin_prefetch(nbr + ((size_t)b * NPTS + n + 1) * KNN, 0, 0);

        // gather neighbor `row`, center-relative coords (xyz is [B,3,N])
        const int nb = nbr[((size_t)b * NPTS + n) * KNN + row];
        const float r0 = px[nb]            - px[n];
        const float r1 = px[NPTS + nb]     - px[NPTS + n];
        const float r2 = px[2*NPTS + nb]   - px[2*NPTS + n];

        // ---- layer 0 (3 -> 64) in VALU; lane covers cols [half*32, +32) ----
#pragma unroll
        for (int c = 0; c < 32; c += 2) {
            const int o = (half << 5) + c;
            float ha = fmaf(r2, W0[o*3+2], fmaf(r1, W0[o*3+1], fmaf(r0, W0[o*3+0], b0[o])));
            float hb = fmaf(r2, W0[o*3+5], fmaf(r1, W0[o*3+4], fmaf(r0, W0[o*3+3], b0[o+1])));
            ha = fmaxf(ha, 0.0f);
            hb = fmaxf(hb, 0.0f);
            const _Float16 f0 = (_Float16)ha, f1 = (_Float16)hb;
            const unsigned pk = (unsigned)__builtin_bit_cast(unsigned short, f0) |
                                ((unsigned)__builtin_bit_cast(unsigned short, f1) << 16);
            *(unsigned*)(hrow + row * HSTRIDE + o) = pk;     // ds_store_b32
        }
        wave_lds_sync();                                     // h0 visible wave-wide

        // ---- layer 1 (64 -> 64): 4 N-tiles x 2 K-steps of 16x16x32 WMMA ----
        const v16h a0 = load_afrag(hrow, row, half, 0);
        const v16h a1 = load_afrag(hrow, row, half, 32);
        v8f acc1[4];
#pragma unroll
        for (int j = 0; j < 4; ++j) {
            const float bias = b1[(j << 4) + (lane & 15)];
            v8f acc;
#pragma unroll
            for (int r = 0; r < 8; ++r) acc[r] = bias;
            const v16h bb0 = load_bfrag(sW1, j * 2 + 0, lane);
            const v16h bb1 = load_bfrag(sW1, j * 2 + 1, lane);
            acc = __builtin_amdgcn_wmma_f32_16x16x32_f16(false, a0, false, bb0,
                                                         (short)0, acc, false, false);
            acc = __builtin_amdgcn_wmma_f32_16x16x32_f16(false, a1, false, bb1,
                                                         (short)0, acc, false, false);
            acc1[j] = acc;
        }
        wave_lds_sync();                                     // a-frags consumed

        // store h1 (ReLU, f16): D-tile (r,lane) -> row half*8+r, col j*16+(lane&15)
#pragma unroll
        for (int j = 0; j < 4; ++j) {
#pragma unroll
            for (int r = 0; r < 8; ++r) {
                const int rr = (half << 3) + r;
                hrow[rr * HSTRIDE + (j << 4) + (lane & 15)] =
                    (_Float16)fmaxf(acc1[j][r], 0.0f);
            }
        }
        wave_lds_sync();                                     // h1 visible wave-wide

        // ---- layer 2 (64 -> 128) + ReLU + max over the 16 neighbor rows ----
        const v16h c0 = load_afrag(hrow, row, half, 0);
        const v16h c1 = load_afrag(hrow, row, half, 32);
#pragma unroll
        for (int j = 0; j < 8; ++j) {
            const float bias = b2[(j << 4) + (lane & 15)];
            v8f acc;
#pragma unroll
            for (int r = 0; r < 8; ++r) acc[r] = bias;
            const v16h bb0 = load_bfrag(sW2, j * 2 + 0, lane);
            const v16h bb1 = load_bfrag(sW2, j * 2 + 1, lane);
            acc = __builtin_amdgcn_wmma_f32_16x16x32_f16(false, c0, false, bb0,
                                                         (short)0, acc, false, false);
            acc = __builtin_amdgcn_wmma_f32_16x16x32_f16(false, c1, false, bb1,
                                                         (short)0, acc, false, false);
            // ReLU then max over rows 0..7 (this lane half), merge halves via xor-16
            float mx = fmaxf(acc[0], 0.0f);
#pragma unroll
            for (int r = 1; r < 8; ++r) mx = fmaxf(mx, fmaxf(acc[r], 0.0f));
            mx = fmaxf(mx, __shfl_xor(mx, 16, 32));
            if (half == 0) {
                const int oc = (j << 4) + lane;              // output channel
                oT[(oc << 6) + (wave << 3) + p] = (_Float16)mx;
            }
        }
        wave_lds_sync();                                     // WAR: next p rewrites hAct
    }

    // coalesced block writeback: [128 oc][64 consecutive n] tile
    __syncthreads();
    const int ob = blockIdx.x >> 6;                          // batch of this block
    const int n0 = (blockIdx.x & 63) << 6;                   // first point of tile
    for (int e = tid; e < 128 * 64; e += MLPB) {
        const int oc = e >> 6;
        const int cn = e & 63;
        out[((size_t)ob * 128 + oc) * NPTS + n0 + cn] = (float)oT[e];
    }
}

// ------------------------------ host launcher -------------------------------
extern "C" void kernel_launch(void* const* d_in, const int* in_sizes, int n_in,
                              void* d_out, int out_size, void* d_ws, size_t ws_size,
                              hipStream_t stream) {
    (void)in_sizes; (void)n_in; (void)out_size; (void)ws_size;
    const float* xyz = (const float*)d_in[0];
    const float* W0  = (const float*)d_in[1];
    const float* b0  = (const float*)d_in[2];
    const float* W1  = (const float*)d_in[3];
    const float* b1  = (const float*)d_in[4];
    const float* W2  = (const float*)d_in[5];
    const float* b2  = (const float*)d_in[6];
    float* out = (float*)d_out;
    int*   idx = (int*)d_ws;                 // B*N*K ints = 2 MB scratch

    knn_kernel<<<256, 128, 0, stream>>>(xyz, idx);
    edge_mlp_kernel<<<512, MLPB, 0, stream>>>(xyz, W0, b0, W1, b1, W2, b2, idx, out);
}
